// Fullmodel_8650064134364
// MI455X (gfx1250) — matrix operand order repaired and checked by reading, not compiled, not run on hardware
//
#include <hip/hip_runtime.h>
#include <hip/hip_bf16.h>

// ---------------------------------------------------------------------------
// Fused FourierFeatures + SIREN MLP for MI455X (gfx1250), wave32 + WMMA.
//
// Shapes: N=262144 points, IN=3, FF=256 (128 freqs), H=256, 4 hidden sine
// layers, OUT=2.  All 5 H x H GEMMs run on v_wmma_f32_16x16x32_f16 with
// activations resident in LDS (f16), so HBM traffic is just x + out + weights.
// ---------------------------------------------------------------------------

#define NPTS   262144
#define HD     256          // hidden width == fourier feature width
#define NFREQ  128
#define TM     128          // rows (points) per workgroup
#define LSTR   264          // LDS row stride in f16 elements (16B aligned, padded)
#define NLAYER 5            // 1 first SIREN layer + 4 hidden sine layers
#define OMEGA0 30.0f

typedef _Float16 v16h __attribute__((ext_vector_type(16)));
typedef _Float16 h8   __attribute__((ext_vector_type(8)));
typedef float    v8f  __attribute__((ext_vector_type(8)));

// ---------------------------------------------------------------------------
// Prep: convert f32 weights (W0 then Wh[0..3]) to f16 into workspace.
// Layout: w16[layer][n][k], row-major over n (output neuron), k (input).
// ---------------------------------------------------------------------------
__global__ __launch_bounds__(256) void convert_weights(
    const float* __restrict__ W0, const float* __restrict__ Wh,
    _Float16* __restrict__ w16)
{
    int i = blockIdx.x * blockDim.x + threadIdx.x;     // 0 .. 5*65536-1
    float v = (i < HD * HD) ? W0[i] : Wh[i - HD * HD];
    w16[i] = (_Float16)v;
}

// ---------------------------------------------------------------------------
// Main fused kernel. One block = 128 rows, 8 waves; each wave owns 16 rows
// end-to-end (no cross-wave LDS sharing -> no barriers needed).
// ---------------------------------------------------------------------------
__global__ __launch_bounds__(256) void siren_fused_kernel(
    const float* __restrict__ x,       // [N,3]
    const float* __restrict__ scales,  // [1,3]
    const float* __restrict__ Bm,      // [3,128]
    const float* __restrict__ b0,      // [256]
    const float* __restrict__ bh,      // [4,256]
    const float* __restrict__ Wf,      // [2,256]
    const float* __restrict__ bf,      // [2]
    const _Float16* __restrict__ Wk,   // [5,256,256] f16 (converted)
    float* __restrict__ out)           // [N,2]
{
    extern __shared__ _Float16 smem[];
    _Float16* act0 = smem;                 // [TM][LSTR]
    _Float16* act1 = smem + TM * LSTR;     // [TM][LSTR]

    const int wave = threadIdx.x >> 5;     // 0..7
    const int lane = threadIdx.x & 31;     // wave32
    const int m0   = wave * 16;            // first row owned by this wave
    const int rowbase = blockIdx.x * TM;   // global row offset of this block

    const int halfsel = lane >> 4;         // 0: lanes 0-15, 1: lanes 16-31
    const int lr      = lane & 15;

    // ------------------------------------------------------------------
    // Fourier feature prologue: h[:, f] = sqrt2*sin(p), h[:, f+128] = sqrt2*cos(p)
    // p = (x*scales) @ B.  Each lane covers 4 of the 128 frequencies.
    // ------------------------------------------------------------------
    {
        const float s0 = scales[0], s1 = scales[1], s2 = scales[2];
        const float sqrt2 = 1.41421356237309515f;
        #pragma unroll 4
        for (int r = 0; r < 16; ++r) {
            const int g = (rowbase + m0 + r) * 3;
            const float x0 = x[g + 0] * s0;
            const float x1 = x[g + 1] * s1;
            const float x2 = x[g + 2] * s2;
            _Float16* hrow = act0 + (m0 + r) * LSTR;
            #pragma unroll
            for (int fi = 0; fi < 4; ++fi) {
                const int f = lane + fi * 32;
                const float p = x0 * Bm[0 * NFREQ + f]
                              + x1 * Bm[1 * NFREQ + f]
                              + x2 * Bm[2 * NFREQ + f];
                hrow[f]         = (_Float16)(__sinf(p) * sqrt2);
                hrow[f + NFREQ] = (_Float16)(__cosf(p) * sqrt2);
            }
        }
    }

    // ------------------------------------------------------------------
    // 5 fused sine layers on WMMA:  h <- sin(omega * (h @ W^T + bias))
    // A = activation tile (M=16 x K=32 f16), B = W^T tile (K=32 x N=16 f16),
    // C/D = f32 16x16 accumulator; bias folded into C init.
    // ------------------------------------------------------------------
    _Float16* cur = act0;
    _Float16* nxt = act1;

    for (int layer = 0; layer < NLAYER; ++layer) {
        const _Float16* __restrict__ W = Wk + layer * HD * HD;  // W[n][k]
        const float* __restrict__ bias = (layer == 0) ? b0 : (bh + (layer - 1) * HD);
        const float omega = (layer == 0) ? OMEGA0 : 1.0f;

        // Warm next layer's weights into cache (global_prefetch_b8 path).
        if (layer + 1 < NLAYER)
            __builtin_prefetch(Wk + (layer + 1) * HD * HD + (threadIdx.x << 6), 0, 1);

        // A-fragment source row for this lane (this wave's own rows).
        const _Float16* __restrict__ arow = cur + (m0 + lr) * LSTR;

        for (int nt = 0; nt < 16; ++nt) {          // 16 output-column tiles
            const int n = nt * 16 + lr;            // this lane's output column
            const float bv = bias[n];
            v8f acc = { bv, bv, bv, bv, bv, bv, bv, bv };

            const _Float16* __restrict__ wrow = W + n * HD;   // B source: W[n][k]

            #pragma unroll
            for (int kt = 0; kt < 8; ++kt) {       // K = 256 in steps of 32
                const int k0 = kt * 32;
                // A 16x32 f16: lane<16 holds K k0..k0+15;
                // lane>=16 holds K k0+16..k0+31 (same k order as B).
                const int ka = k0 + halfsel * 16;
                h8 alo = *(const h8*)(arow + ka);
                h8 ahi = *(const h8*)(arow + ka + 8);
                v16h a = __builtin_shufflevector(alo, ahi,
                    0, 1, 2, 3, 4, 5, 6, 7, 8, 9, 10, 11, 12, 13, 14, 15);
                // B 32x16 f16: lanes 0-15 hold K k0..k0+15 of column n,
                // lanes 16-31 hold K k0+16..k0+31 (contiguous in W[n][k]).
                const int kb = k0 + halfsel * 16;
                v16h b = *(const v16h*)(wrow + kb);

                acc = __builtin_amdgcn_wmma_f32_16x16x32_f16(
                    /*neg_a=*/false, a, /*neg_b=*/false, b,
                    /*c_mod=*/(short)0, acc,
                    /*reuse_a=*/false, /*reuse_b=*/false);
            }

            // D layout: VGPR r -> row m0 + r + 8*halfsel, column n.
            _Float16* __restrict__ drow = nxt + (m0 + 8 * halfsel) * LSTR + n;
            #pragma unroll
            for (int r = 0; r < 8; ++r)
                drow[r * LSTR] = (_Float16)__sinf(omega * acc[r]);
        }

        _Float16* t = cur; cur = nxt; nxt = t;     // ping-pong LDS buffers
    }

    // ------------------------------------------------------------------
    // Head: out = h @ Wf^T + bf  (H=256 -> OUT=2).  32 lanes = 16 rows x 2 outs.
    // ------------------------------------------------------------------
    {
        const int r = lr;                  // row within wave tile
        const int j = halfsel;             // output index 0/1
        const _Float16* __restrict__ hrow = cur + (m0 + r) * LSTR;
        const float* __restrict__ wf = Wf + j * HD;
        float accv = bf[j];
        #pragma unroll 8
        for (int k = 0; k < HD; ++k)
            accv += (float)hrow[k] * wf[k];
        out[(rowbase + m0 + r) * 2 + j] = accv;
    }
}

// ---------------------------------------------------------------------------
// Launch wrapper.  Inputs (setup_inputs order):
//   0:x [N,3] f32   1:scales [1,3]   2:B [3,128]   3:W0 [256,256]
//   4:b0 [256]      5:Wh [4,256,256] 6:bh [4,256]  7:Wf [2,256]   8:bf [2]
// d_out: [N,2] f32.  d_ws: >= 5*256*256*2 bytes (655,360 B) for f16 weights.
// ---------------------------------------------------------------------------
extern "C" void kernel_launch(void* const* d_in, const int* in_sizes, int n_in,
                              void* d_out, int out_size, void* d_ws, size_t ws_size,
                              hipStream_t stream)
{
    const float* x      = (const float*)d_in[0];
    const float* scales = (const float*)d_in[1];
    const float* Bm     = (const float*)d_in[2];
    const float* W0     = (const float*)d_in[3];
    const float* b0     = (const float*)d_in[4];
    const float* Wh     = (const float*)d_in[5];
    const float* bh     = (const float*)d_in[6];
    const float* Wf     = (const float*)d_in[7];
    const float* bf     = (const float*)d_in[8];
    float*       out    = (float*)d_out;
    _Float16*    w16    = (_Float16*)d_ws;

    // 1) Convert 5 layers of f32 weights to f16 in workspace.
    const int wtotal = NLAYER * HD * HD;                 // 327,680 elements
    convert_weights<<<wtotal / 256, 256, 0, stream>>>(W0, Wh, w16);

    // 2) Fused network: 2048 blocks x 256 threads, 132 KB dynamic LDS each.
    const size_t shmem = (size_t)2 * TM * LSTR * sizeof(_Float16);
    siren_fused_kernel<<<NPTS / TM, 256, shmem, stream>>>(
        x, scales, Bm, b0, bh, Wf, bf, w16, out);
}
